// EfficientWindowAttention_67164698575295
// MI455X (gfx1250) — compile-verified, matrix operand order checked by reading, not run on hardware
//
#include <hip/hip_runtime.h>
#include <hip/hip_bf16.h>

typedef __bf16 bf16;
typedef __attribute__((ext_vector_type(16))) __bf16 v16bf;
typedef __attribute__((ext_vector_type(8)))  __bf16 v8bf;
typedef __attribute__((ext_vector_type(4)))  __bf16 v4bf;
typedef __attribute__((ext_vector_type(2)))  __bf16 v2bf;
typedef __attribute__((ext_vector_type(8)))  float   v8f;
typedef __attribute__((ext_vector_type(4)))  unsigned int v4u;
typedef __attribute__((ext_vector_type(8)))  int v8i;
typedef __attribute__((ext_vector_type(4)))  int v4i;

#define C_DIM   512
#define N_TOK   64
#define NHEADS  16
#define HDIM    32
#define B_WIN   2048
#define XQ_STRIDE 520   // padded row stride (bf16) for x/q/k: 260 dwords = 4 mod 64 banks
#define V_STRIDE  72    // padded row stride (bf16) for v^T: 9 x 16B groups, conflict-free
#define QK_SCALE 0.17677669529663687f

#ifndef __has_builtin
#define __has_builtin(x) 0
#endif
#if __has_builtin(__builtin_amdgcn_tensor_load_to_lds) && __has_builtin(__builtin_amdgcn_s_wait_tensorcnt)
#define USE_TDM 1
#else
#define USE_TDM 0
#endif

// ---------------------------------------------------------------------------
// 16-bit WMMA fragment loader (A and B share the per-lane pattern): lane l
// holds row (row0 + l%16); lanes 0-15 carry K chunks [kb..kb+7],[kb+16..kb+23];
// lanes 16-31 carry [kb+8..kb+15],[kb+24..kb+31]. 2 x 16-byte loads.
// ---------------------------------------------------------------------------
__device__ __forceinline__ v16bf load_frag(const bf16* base, int row0, int stride,
                                           int kb, int lane) {
  const int r    = row0 + (lane & 15);
  const int koff = (lane >> 4) << 3;
  const bf16* p  = base + (size_t)r * stride + kb + koff;
  v8bf lo = *reinterpret_cast<const v8bf*>(p);
  v8bf hi = *reinterpret_cast<const v8bf*>(p + 16);
  v16bf out;
#pragma unroll
  for (int i = 0; i < 8; ++i) { out[i] = lo[i]; out[i + 8] = hi[i]; }
  return out;
}

// Fragment loader for the XOR-swizzled P buffer ([64][64], 8-elem chunk ^ (row&7)).
__device__ __forceinline__ v16bf load_frag_p(const bf16* base, int row0, int kb, int lane) {
  const int r  = row0 + (lane & 15);
  const int ch = lane >> 4;
  const int c0 = ((kb >> 3) + ch)     ^ (r & 7);
  const int c1 = ((kb >> 3) + ch + 2) ^ (r & 7);
  v8bf lo = *reinterpret_cast<const v8bf*>(base + r * 64 + (c0 << 3));
  v8bf hi = *reinterpret_cast<const v8bf*>(base + r * 64 + (c1 << 3));
  v16bf out;
#pragma unroll
  for (int i = 0; i < 8; ++i) { out[i] = lo[i]; out[i + 8] = hi[i]; }
  return out;
}

__device__ __forceinline__ v8f wmma_bf16(v16bf a, v16bf b, v8f c) {
  return __builtin_amdgcn_wmma_f32_16x16x32_bf16(false, a, false, b,
                                                 (short)0, c, false, false);
}

#if USE_TDM
// Issue one TDM descriptor: 1-D contiguous copy, global -> LDS (TENSORcnt).
__device__ __forceinline__ void tdm_load_1d(unsigned int lds_addr, const void* gptr,
                                            unsigned int n_elems_f32) {
  unsigned long long ga = (unsigned long long)(uintptr_t)gptr;
  v4u g0 = { 1u,                                   // count=1 (valid user descriptor)
             lds_addr,                             // D#.lds_addr
             (unsigned int)(ga & 0xffffffffu),     // global_addr[31:0]
             (unsigned int)((ga >> 32) & 0x01ffffffu) | 0x80000000u }; // [56:32] | type=2
  v8i g1 = { (int)0x00020000,                      // wg_mask=0, data_size=2 (4B)
             (int)((n_elems_f32 & 0xffffu) << 16), // tensor_dim0[15:0]
             (int)((n_elems_f32 >> 16) | (1u << 16)),        // td0[31:16] | tensor_dim1=1
             (int)((n_elems_f32 & 0xffffu) << 16), // tile_dim0 (fits u16: 32768)
             0,                                    // tile_dim1=0, tile_dim2=0
             (int)n_elems_f32,                     // tensor_dim0_stride[31:0]
             0, 0 };
  v4i z4 = {0, 0, 0, 0};
#if __clang_major__ >= 23
  v8i z8 = {0, 0, 0, 0, 0, 0, 0, 0};
  __builtin_amdgcn_tensor_load_to_lds(g0, g1, z4, z4, z8, 0);
#else
  __builtin_amdgcn_tensor_load_to_lds(g0, g1, z4, z4, 0);
#endif
}
#endif

// ---------------------------------------------------------------------------
// Kernel 0: weight fp32->bf16 conversion + dense bias gather bias[h][n][m]
// ---------------------------------------------------------------------------
__global__ __launch_bounds__(256)
void prep_kernel(const float* __restrict__ qkv_w, const float* __restrict__ proj_w,
                 const float* __restrict__ bias_table, const int* __restrict__ rel_index,
                 bf16* __restrict__ wqkv, bf16* __restrict__ wproj,
                 float* __restrict__ biasf) {
  const int t = blockIdx.x * 256 + threadIdx.x;
  if (t < 3 * C_DIM * C_DIM) wqkv[t] = (bf16)qkv_w[t];
  if (t < C_DIM * C_DIM)     wproj[t] = (bf16)proj_w[t];
  if (t < NHEADS * N_TOK * N_TOK) {
    const int h  = t >> 12;
    const int nm = t & 4095;
    biasf[t] = bias_table[rel_index[nm] * NHEADS + h];
  }
}

// ---------------------------------------------------------------------------
// Kernel 1: fused QKV projection + windowed attention. 1 block = 1 window.
// 256 threads = 8 waves. LDS (273,408 B): padded x (reused as swizzled P),
// padded q/k (staging target of the TDM fp32 x DMA), padded v^T.
// ---------------------------------------------------------------------------
__global__ __launch_bounds__(256)
void win_attn_kernel(const float* __restrict__ x, const bf16* __restrict__ wqkv,
                     const float* __restrict__ qkv_b, const float* __restrict__ biasf,
                     bf16* __restrict__ attn_out) {
  __shared__ bf16 s_x[N_TOK * XQ_STRIDE];       // 66,560 B; later: swizzled P buffers
  __shared__ bf16 s_qk[2 * N_TOK * XQ_STRIDE];  // 133,120 B; stage0: fp32 x staging
  __shared__ bf16 s_v[C_DIM * V_STRIDE];        // 73,728 B; v transposed [512][64+8]

  bf16* s_q = s_qk;
  bf16* s_k = s_qk + N_TOK * XQ_STRIDE;

  const int b    = blockIdx.x;
  const int tid  = threadIdx.x;
  const int wv   = tid >> 5;
  const int ln   = tid & 31;
  const int nn   = ln & 15;
  const int half = ln >> 4;

  // ---- Stage 0: x[b] (fp32, 128 KB contiguous) -> bf16 padded LDS ----
#if USE_TDM
  if (wv == 0) {
    tdm_load_1d((unsigned int)(uintptr_t)(void*)s_qk,
                (const void*)(x + (size_t)b * N_TOK * C_DIM),
                (unsigned int)(N_TOK * C_DIM));
    __builtin_amdgcn_s_wait_tensorcnt(0);
  }
  __syncthreads();
  {
    const float4* sb = reinterpret_cast<const float4*>((const void*)s_qk);
    for (int t = tid; t < (N_TOK * C_DIM) / 4; t += 256) {
      float4 v4 = sb[t];
      const int r = t >> 7;
      const int c = (t & 127) << 2;
      v4bf pk; pk[0] = (bf16)v4.x; pk[1] = (bf16)v4.y; pk[2] = (bf16)v4.z; pk[3] = (bf16)v4.w;
      *reinterpret_cast<v4bf*>(s_x + r * XQ_STRIDE + c) = pk;
    }
  }
#else
  {
    const float4* xb = reinterpret_cast<const float4*>(x + (size_t)b * N_TOK * C_DIM);
    for (int t = tid; t < (N_TOK * C_DIM) / 4; t += 256) {
      float4 v4 = xb[t];
      const int r = t >> 7;
      const int c = (t & 127) << 2;
      v4bf pk; pk[0] = (bf16)v4.x; pk[1] = (bf16)v4.y; pk[2] = (bf16)v4.z; pk[3] = (bf16)v4.w;
      *reinterpret_cast<v4bf*>(s_x + r * XQ_STRIDE + c) = pk;
    }
  }
#endif
  __syncthreads();

  // ---- Stage 1: QKV GEMM [64,512] x [512,1536]^T; wave w owns 192 out cols ----
  for (int nt = 0; nt < 12; ++nt) {
    const int col0 = wv * 192 + nt * 16;
    v8f acc[4];
#pragma unroll
    for (int i = 0; i < 4; ++i)
#pragma unroll
      for (int e = 0; e < 8; ++e) acc[i][e] = 0.0f;

    for (int ks = 0; ks < 16; ++ks) {
      const int kb = ks << 5;
      v16bf wfrag = load_frag(wqkv, col0, C_DIM, kb, ln);   // B: weight rows = out cols
#pragma unroll
      for (int i = 0; i < 4; ++i) {
        v16bf af = load_frag(s_x, i * 16, XQ_STRIDE, kb, ln);
        acc[i] = wmma_bf16(af, wfrag, acc[i]);
      }
    }
    const int   col  = col0 + nn;
    const float bias = qkv_b[col];
    const int   mat  = col >> 9;         // 0:q 1:k 2:v (uniform per 16-col tile)
    const int   rem  = col & 511;        // h*32 + d
    if (mat == 2) {                      // v: transposed rows, pack pairs -> b32 stores
#pragma unroll
      for (int i = 0; i < 4; ++i)
#pragma unroll
        for (int r = 0; r < 8; r += 2) {
          const int m = i * 16 + half * 8 + r;
          v2bf pk;
          pk[0] = (bf16)(acc[i][r]     + bias);
          pk[1] = (bf16)(acc[i][r + 1] + bias);
          *reinterpret_cast<v2bf*>(s_v + rem * V_STRIDE + m) = pk;
        }
    } else {
      bf16* dst = (mat == 1) ? s_k : s_q;
      const float sc = (mat == 0) ? QK_SCALE : 1.0f;
#pragma unroll
      for (int i = 0; i < 4; ++i)
#pragma unroll
        for (int r = 0; r < 8; ++r) {
          const int m = i * 16 + half * 8 + r;
          dst[m * XQ_STRIDE + rem] = (bf16)((acc[i][r] + bias) * sc);
        }
    }
  }
  __syncthreads();

  // ---- Stage 2: per-head attention; wave wv handles heads wv and wv+8 ----
  bf16* s_p = s_x + wv * (N_TOK * N_TOK);    // per-wave swizzled P (x staging is dead)

  for (int hi = 0; hi < 2; ++hi) {
    const int h = wv + hi * 8;
    const bf16* qh = s_q + h * HDIM;
    const bf16* kh = s_k + h * HDIM;
    const bf16* vh = s_v + h * HDIM * V_STRIDE;

    v16bf qa[4], kf[4];
#pragma unroll
    for (int i = 0; i < 4; ++i) qa[i] = load_frag(qh, i * 16, XQ_STRIDE, 0, ln);
#pragma unroll
    for (int j = 0; j < 4; ++j) kf[j] = load_frag(kh, j * 16, XQ_STRIDE, 0, ln);

    v8f s[4][4];
#pragma unroll
    for (int i = 0; i < 4; ++i)
#pragma unroll
      for (int j = 0; j < 4; ++j) {
#pragma unroll
        for (int e = 0; e < 8; ++e) s[i][j][e] = 0.0f;
        s[i][j] = wmma_bf16(qa[i], kf[j], s[i][j]);
      }

    // relative position bias
    const float* bh = biasf + h * (N_TOK * N_TOK);
#pragma unroll
    for (int i = 0; i < 4; ++i)
#pragma unroll
      for (int j = 0; j < 4; ++j)
#pragma unroll
        for (int r = 0; r < 8; ++r)
          s[i][j][r] += bh[(i * 16 + half * 8 + r) * N_TOK + j * 16 + nn];

    // softmax over N (row = 4 j-tiles x 16 lanes of one half-wave)
#pragma unroll
    for (int i = 0; i < 4; ++i) {
#pragma unroll
      for (int r = 0; r < 8; ++r) {
        float mx = fmaxf(fmaxf(s[i][0][r], s[i][1][r]), fmaxf(s[i][2][r], s[i][3][r]));
        mx = fmaxf(mx, __shfl_xor(mx, 1));
        mx = fmaxf(mx, __shfl_xor(mx, 2));
        mx = fmaxf(mx, __shfl_xor(mx, 4));
        mx = fmaxf(mx, __shfl_xor(mx, 8));
        const float e0 = __expf(s[i][0][r] - mx);
        const float e1 = __expf(s[i][1][r] - mx);
        const float e2 = __expf(s[i][2][r] - mx);
        const float e3 = __expf(s[i][3][r] - mx);
        float sum = (e0 + e1) + (e2 + e3);
        sum += __shfl_xor(sum, 1);
        sum += __shfl_xor(sum, 2);
        sum += __shfl_xor(sum, 4);
        sum += __shfl_xor(sum, 8);
        const float inv = __builtin_amdgcn_rcpf(sum);
        s[i][0][r] = e0 * inv; s[i][1][r] = e1 * inv;
        s[i][2][r] = e2 * inv; s[i][3][r] = e3 * inv;
      }
    }

    // P -> per-wave LDS, XOR-swizzled chunks (conflict-free A-fragment reload)
#pragma unroll
    for (int i = 0; i < 4; ++i)
#pragma unroll
      for (int j = 0; j < 4; ++j)
#pragma unroll
        for (int r = 0; r < 8; ++r) {
          const int M   = i * 16 + half * 8 + r;
          const int col = j * 16 + nn;
          const int cs  = (col >> 3) ^ (M & 7);
          s_p[M * 64 + (cs << 3) + (col & 7)] = (bf16)s[i][j][r];
        }

    // O = P @ V : [64,64] x [64,32], K split into 2 chunks of 32
    v8f o[4][2];
#pragma unroll
    for (int i = 0; i < 4; ++i)
#pragma unroll
      for (int j = 0; j < 2; ++j)
#pragma unroll
        for (int e = 0; e < 8; ++e) o[i][j][e] = 0.0f;

#pragma unroll
    for (int ks = 0; ks < 2; ++ks) {
      const int kb = ks * 32;
      v16bf vb0 = load_frag(vh, 0,  V_STRIDE, kb, ln);  // B rows = d (v transposed)
      v16bf vb1 = load_frag(vh, 16, V_STRIDE, kb, ln);
#pragma unroll
      for (int i = 0; i < 4; ++i) {
        v16bf pa = load_frag_p(s_p, i * 16, kb, ln);
        o[i][0] = wmma_bf16(pa, vb0, o[i][0]);
        o[i][1] = wmma_bf16(pa, vb1, o[i][1]);
      }
    }

    // store head output to ws as bf16, layout [B*64][512]
    bf16* ao = attn_out + (size_t)b * N_TOK * C_DIM + h * HDIM;
#pragma unroll
    for (int i = 0; i < 4; ++i)
#pragma unroll
      for (int j = 0; j < 2; ++j)
#pragma unroll
        for (int r = 0; r < 8; ++r)
          ao[(size_t)(i * 16 + half * 8 + r) * C_DIM + j * 16 + nn] = (bf16)o[i][j][r];
  }
}

// ---------------------------------------------------------------------------
// Kernel 2: output projection [131072,512] x [512,512]^T + bias -> fp32 out.
// Block tile 128x128; 8 waves as 4(M) x 2(N); wave tile 32x64; no LDS needed.
// ---------------------------------------------------------------------------
__global__ __launch_bounds__(256)
void proj_kernel(const bf16* __restrict__ A, const bf16* __restrict__ W,
                 const float* __restrict__ pb, float* __restrict__ out) {
  const int tid  = threadIdx.x;
  const int wv   = tid >> 5;
  const int ln   = tid & 31;
  const int nn   = ln & 15;
  const int half = ln >> 4;
  const int row0 = blockIdx.x * 128 + (wv & 3) * 32;
  const int col0 = blockIdx.y * 128 + (wv >> 2) * 64;

  v8f acc[2][4];
#pragma unroll
  for (int i = 0; i < 2; ++i)
#pragma unroll
    for (int j = 0; j < 4; ++j)
#pragma unroll
      for (int e = 0; e < 8; ++e) acc[i][j][e] = 0.0f;

  for (int ks = 0; ks < 16; ++ks) {
    const int kb = ks << 5;
    if (ks + 1 < 16) {  // prefetch next K chunk of the streamed A rows
      __builtin_prefetch((const void*)(A + (size_t)(row0 + nn) * C_DIM + kb + 32), 0, 1);
      __builtin_prefetch((const void*)(A + (size_t)(row0 + 16 + nn) * C_DIM + kb + 32), 0, 1);
    }
    v16bf a0 = load_frag(A, row0,      C_DIM, kb, ln);
    v16bf a1 = load_frag(A, row0 + 16, C_DIM, kb, ln);
#pragma unroll
    for (int j = 0; j < 4; ++j) {
      v16bf bw = load_frag(W, col0 + j * 16, C_DIM, kb, ln);
      acc[0][j] = wmma_bf16(a0, bw, acc[0][j]);
      acc[1][j] = wmma_bf16(a1, bw, acc[1][j]);
    }
  }
#pragma unroll
  for (int i = 0; i < 2; ++i)
#pragma unroll
    for (int j = 0; j < 4; ++j) {
      const int   colc = col0 + j * 16 + nn;
      const float bias = pb[colc];
#pragma unroll
      for (int r = 0; r < 8; ++r) {
        const int row = row0 + i * 16 + half * 8 + r;
        out[(size_t)row * C_DIM + colc] = acc[i][j][r] + bias;
      }
    }
}

// ---------------------------------------------------------------------------
extern "C" void kernel_launch(void* const* d_in, const int* in_sizes, int n_in,
                              void* d_out, int out_size, void* d_ws, size_t ws_size,
                              hipStream_t stream) {
  const float* x          = (const float*)d_in[0];
  const float* qkv_w      = (const float*)d_in[1];
  const float* qkv_b      = (const float*)d_in[2];
  const float* proj_w     = (const float*)d_in[3];
  const float* proj_b     = (const float*)d_in[4];
  const float* bias_table = (const float*)d_in[5];
  const int*   rel_index  = (const int*)d_in[6];
  float* out = (float*)d_out;

  char* ws = (char*)d_ws;
  bf16*  wqkv  = (bf16*)(ws);                 // 3*512*512*2   = 1,572,864 B
  bf16*  wproj = (bf16*)(ws + 1572864);       //   512*512*2   =   524,288 B
  float* biasf = (float*)(ws + 2097152);      // 16*64*64*4    =   262,144 B
  bf16*  attn  = (bf16*)(ws + 2359296);       // 2048*64*512*2 = 134,217,728 B

  hipLaunchKernelGGL(prep_kernel, dim3(3072), dim3(256), 0, stream,
                     qkv_w, proj_w, bias_table, rel_index, wqkv, wproj, biasf);
  hipLaunchKernelGGL(win_attn_kernel, dim3(B_WIN), dim3(256), 0, stream,
                     x, wqkv, qkv_b, biasf, attn);
  hipLaunchKernelGGL(proj_kernel, dim3(1024, 4), dim3(256), 0, stream,
                     attn, wproj, proj_b, out);
}
